// ConformerLayer_15582141350307
// MI455X (gfx1250) — compile-verified, matrix operand order checked by reading, not run on hardware
//
#include <hip/hip_runtime.h>

// ---------------------------------------------------------------------------
// Conformer layer forward for gfx1250 (MI455X).
// All GEMMs run on v_wmma_f32_16x16x32_f16 (f16 operands, fp32 accum),
// staged through LDS with GLOBAL_LOAD_ASYNC_TO_LDS_B128 (ASYNCcnt-tracked).
// Wave32, 4 waves / block, 64x64 block tile, K-step 64 (8 WMMA per stage).
// ---------------------------------------------------------------------------

typedef __attribute__((ext_vector_type(16))) _Float16 v16h;
typedef __attribute__((ext_vector_type(8)))  _Float16 v8h;
typedef __attribute__((ext_vector_type(8)))  float    v8f;

__device__ __forceinline__ float conf_sigmoid(float x) {
    return 1.0f / (1.0f + __expf(-x));
}

__device__ __forceinline__ void conf_wait_async0() {
#if defined(__has_builtin)
#if __has_builtin(__builtin_amdgcn_s_wait_asynccnt)
    __builtin_amdgcn_s_wait_asynccnt(0);
    return;
#else
    asm volatile("s_wait_asynccnt 0x0" ::: "memory");
    return;
#endif
#else
    asm volatile("s_wait_asynccnt 0x0" ::: "memory");
#endif
}

// async memory -> LDS copy of 16 bytes (one lane)
__device__ __forceinline__ void conf_async_b128(unsigned lds_off,
                                                const _Float16* src) {
    asm volatile("global_load_async_to_lds_b128 %0, %1, off"
                 :: "v"(lds_off), "v"(src) : "memory");
}

// ---------------------------------------------------------------------------
// LayerNorm over last dim (one block per row, 128 threads)
// ---------------------------------------------------------------------------
__global__ __launch_bounds__(128) void conf_ln(
    const float* __restrict__ x, const float* __restrict__ g,
    const float* __restrict__ b, float* __restrict__ y, int D)
{
    const int row = blockIdx.x;
    const int tid = threadIdx.x;
    const float* xr = x + (size_t)row * D;
    float s = 0.f, s2 = 0.f;
    for (int j = tid; j < D; j += 128) {
        float v = xr[j];
        s += v; s2 += v * v;
    }
    __shared__ float sh1[128], sh2[128];
    sh1[tid] = s; sh2[tid] = s2;
    __syncthreads();
    for (int st = 64; st > 0; st >>= 1) {
        if (tid < st) { sh1[tid] += sh1[tid + st]; sh2[tid] += sh2[tid + st]; }
        __syncthreads();
    }
    __shared__ float mu, inv;
    if (tid == 0) {
        float m = sh1[0] / (float)D;
        float v = sh2[0] / (float)D - m * m;
        mu = m; inv = rsqrtf(v + 1e-5f);
    }
    __syncthreads();
    float* yr = y + (size_t)row * D;
    for (int j = tid; j < D; j += 128)
        yr[j] = (xr[j] - mu) * inv * g[j] + b[j];
}

// ---------------------------------------------------------------------------
// fp32 -> f16 convert
// ---------------------------------------------------------------------------
__global__ __launch_bounds__(256) void conf_cvt(
    const float* __restrict__ in, _Float16* __restrict__ out, long n)
{
    long i = (long)blockIdx.x * 256 + threadIdx.x;
    if (i < n) out[i] = (_Float16)in[i];
}

// q + per-head bias (pbu/pbv broadcast over rows), convert to f16
__global__ __launch_bounds__(256) void conf_bias_head(
    const float* __restrict__ q, const float* __restrict__ bias,
    _Float16* __restrict__ out, long n, int D)
{
    long i = (long)blockIdx.x * 256 + threadIdx.x;
    if (i < n) out[i] = (_Float16)(q[i] + bias[i % D]);
}

// GLU over channels: out[row,c] = a * sigmoid(g), in is [rows, 2D]
__global__ __launch_bounds__(256) void conf_glu(
    const float* __restrict__ in, float* __restrict__ out, long nOut, int D)
{
    long i = (long)blockIdx.x * 256 + threadIdx.x;
    if (i < nOut) {
        long row = i / D; int c = (int)(i - row * D);
        float a = in[row * (2 * (long)D) + c];
        float gg = in[row * (2 * (long)D) + D + c];
        out[i] = a * conf_sigmoid(gg);
    }
}

__global__ __launch_bounds__(256) void conf_silu(float* __restrict__ x, long n)
{
    long i = (long)blockIdx.x * 256 + threadIdx.x;
    if (i < n) { float v = x[i]; x[i] = v * conf_sigmoid(v); }
}

// r += a*y
__global__ __launch_bounds__(256) void conf_axpy(
    float* __restrict__ r, const float* __restrict__ y, float a, long n)
{
    long i = (long)blockIdx.x * 256 + threadIdx.x;
    if (i < n) r[i] += a * y[i];
}

// r = x + a*y
__global__ __launch_bounds__(256) void conf_add3(
    float* __restrict__ r, const float* __restrict__ x,
    const float* __restrict__ y, float a, long n)
{
    long i = (long)blockIdx.x * 256 + threadIdx.x;
    if (i < n) r[i] = x[i] + a * y[i];
}

// depthwise conv1d over time, kernel K, pad (K-1)/2, + bias
__global__ __launch_bounds__(256) void conf_dwconv(
    const float* __restrict__ in, const float* __restrict__ wgt,
    const float* __restrict__ bias, float* __restrict__ out,
    int Bn, int T, int D, int K)
{
    long i = (long)blockIdx.x * 256 + threadIdx.x;
    long n = (long)Bn * T * D;
    if (i >= n) return;
    int d = (int)(i % D);
    int t = (int)((i / D) % T);
    int b = (int)(i / ((long)D * T));
    int pad = (K - 1) / 2;
    float acc = bias[d];
    for (int k = 0; k < K; ++k) {
        int tt = t + k - pad;
        if (tt >= 0 && tt < T)
            acc += in[((size_t)b * T + tt) * D + d] * wgt[d * K + k];
    }
    out[i] = acc;
}

// ---------------------------------------------------------------------------
// softmax with fused NeMo-style relative shift.
// ac: [H,T,T] scores, bd: [H,T,P] rel-pos scores, p = s + (T-1) - t.
// One block per (t,h) row; writes f16 attention probs [H,T,T].
// ---------------------------------------------------------------------------
__global__ __launch_bounds__(256) void conf_softmax_shift(
    const float* __restrict__ ac, const float* __restrict__ bd,
    _Float16* __restrict__ attn, int T, int P, float scale)
{
    const int t = blockIdx.x, h = blockIdx.y, tid = threadIdx.x;
    const float* acr = ac + ((size_t)h * T + t) * T;
    const float* bdr = bd + ((size_t)h * T + t) * P;
    const int sh0 = (T - 1) - t;
    __shared__ float red[256];
    __shared__ float mx, sum;

    float m = -3.0e38f;
    for (int j = tid; j < T; j += 256) {
        float v = (acr[j] + bdr[j + sh0]) * scale;
        m = fmaxf(m, v);
    }
    red[tid] = m; __syncthreads();
    for (int st = 128; st > 0; st >>= 1) {
        if (tid < st) red[tid] = fmaxf(red[tid], red[tid + st]);
        __syncthreads();
    }
    if (tid == 0) mx = red[0];
    __syncthreads();

    float s = 0.f;
    for (int j = tid; j < T; j += 256) {
        float v = (acr[j] + bdr[j + sh0]) * scale;
        s += __expf(v - mx);
    }
    red[tid] = s; __syncthreads();
    for (int st = 128; st > 0; st >>= 1) {
        if (tid < st) red[tid] += red[tid + st];
        __syncthreads();
    }
    if (tid == 0) sum = red[0];
    __syncthreads();

    float rinv = 1.0f / sum;
    _Float16* ar = attn + ((size_t)h * T + t) * T;
    for (int j = tid; j < T; j += 256) {
        float v = (acr[j] + bdr[j + sh0]) * scale;
        ar[j] = (_Float16)(__expf(v - mx) * rinv);
    }
}

// ---------------------------------------------------------------------------
// WMMA GEMM:  C[M,N] = act( A[M,K] * B + bias )
//   A: f16 row-major, lda (elements), per-batch stride sA
//   B: f16; transB=0: row-major [K,N] ldb; transB=1: row-major [N,K] ldb
//   C: fp32, ldc; act: 0=none, 1=SiLU
// Block: 128 threads (4 waves), block tile 64x64, wave tile 32x32 (2x2 WMMA),
// K-step 64 (2 WMMA K-slices per LDS stage -> 8 v_wmma / stage).
// Tile staging uses GLOBAL_LOAD_ASYNC_TO_LDS_B128 on the aligned paths.
// ---------------------------------------------------------------------------
__global__ __launch_bounds__(128) void conf_gemm(
    const _Float16* __restrict__ A, long sA, int lda,
    const _Float16* __restrict__ Bm, long sB, int ldb, int tB,
    float* __restrict__ C, long sC, int ldc,
    int M, int N, int Kd, const float* __restrict__ bias, int act)
{
    constexpr int LSTR = 72;   // 64 halves payload + 8 pad (144B rows, 16B aligned)
    __shared__ alignas(16) _Float16 As[64 * LSTR];
    __shared__ alignas(16) _Float16 Bs[64 * LSTR];

    const int z = blockIdx.z;
    A  += (size_t)z * sA;
    Bm += (size_t)z * sB;
    C  += (size_t)z * sC;

    const int m0 = blockIdx.y * 64, n0 = blockIdx.x * 64;
    const int tid  = threadIdx.x;
    const int wave = tid >> 5, lane = tid & 31;
    const int wr = wave >> 1, wc = wave & 1;   // 2x2 wave grid
    const int lh = lane & 15, hi = lane >> 4;

    v8f c[2][2];
#pragma unroll
    for (int a = 0; a < 2; ++a)
#pragma unroll
        for (int bb = 0; bb < 2; ++bb)
#pragma unroll
            for (int i = 0; i < 8; ++i) c[a][bb][i] = 0.f;

    const int lrow = tid >> 1;          // 0..63 (tile row / tile col)
    const int seg  = (tid & 1) * 32;    // half-row segment (32 halves)
    const bool rowOK = (m0 + lrow) < M;
    const bool colOK = (n0 + lrow) < N;

    // one-time zero fill for out-of-range rows/cols (stays zero all K-steps)
    if (!rowOK) {
        _Float16* dst = &As[lrow * LSTR + seg];
#pragma unroll
        for (int i = 0; i < 32; ++i) dst[i] = (_Float16)0.f;
    }
    if (!colOK) {
        _Float16* dst = &Bs[lrow * LSTR + seg];
#pragma unroll
        for (int i = 0; i < 32; ++i) dst[i] = (_Float16)0.f;
    }

    for (int k0 = 0; k0 < Kd; k0 += 64) {
        // ---- stage A tile: 64 rows x 64 halves, async mem->LDS ----
        if (rowOK) {
            const _Float16* src = A + (size_t)(m0 + lrow) * lda + k0 + seg;
            unsigned lo = (unsigned)(size_t)&As[lrow * LSTR + seg];
#pragma unroll
            for (int i = 0; i < 4; ++i)
                conf_async_b128(lo + i * 16, src + i * 8);
            if (k0 + 64 < Kd) __builtin_prefetch(src + 64, 0, 1);
        }
        // ---- stage B tile column-major in LDS: Bs[col][k] ----
        if (tB) {
            if (colOK) {
                const _Float16* src = Bm + (size_t)(n0 + lrow) * ldb + k0 + seg;
                unsigned lo = (unsigned)(size_t)&Bs[lrow * LSTR + seg];
#pragma unroll
                for (int i = 0; i < 4; ++i)
                    conf_async_b128(lo + i * 16, src + i * 8);
            }
        } else {
            if (colOK) {
                int gc = n0 + lrow;
                _Float16* dst = &Bs[lrow * LSTR + seg];
#pragma unroll
                for (int i = 0; i < 32; ++i)
                    dst[i] = Bm[(size_t)(k0 + seg + i) * ldb + gc];
            }
        }
        conf_wait_async0();
        __syncthreads();

        // ---- 2 K-slices of 32, fragments per ISA 16-bit layout:
        // lanes 0-15: K {0..7, 16..23}; lanes 16-31: K {8..15, 24..31}
#pragma unroll
        for (int kk = 0; kk < 2; ++kk) {
            const int kb = kk * 32;
            v16h af[2], bf[2];
#pragma unroll
            for (int mi = 0; mi < 2; ++mi) {
                int ar = wr * 32 + mi * 16 + lh;
                v8h lo = *(const v8h*)&As[ar * LSTR + kb + hi * 8];
                v8h hv = *(const v8h*)&As[ar * LSTR + kb + 16 + hi * 8];
                af[mi] = __builtin_shufflevector(lo, hv, 0, 1, 2, 3, 4, 5, 6, 7,
                                                 8, 9, 10, 11, 12, 13, 14, 15);
            }
#pragma unroll
            for (int ni = 0; ni < 2; ++ni) {
                int bc = wc * 32 + ni * 16 + lh;
                v8h lo = *(const v8h*)&Bs[bc * LSTR + kb + hi * 8];
                v8h hv = *(const v8h*)&Bs[bc * LSTR + kb + 16 + hi * 8];
                bf[ni] = __builtin_shufflevector(lo, hv, 0, 1, 2, 3, 4, 5, 6, 7,
                                                 8, 9, 10, 11, 12, 13, 14, 15);
            }
#pragma unroll
            for (int mi = 0; mi < 2; ++mi)
#pragma unroll
                for (int ni = 0; ni < 2; ++ni)
                    c[mi][ni] = __builtin_amdgcn_wmma_f32_16x16x32_f16(
                        false, af[mi], false, bf[ni], (short)0, c[mi][ni],
                        false, false);
        }
        __syncthreads();
    }

    // ---- epilogue: C layout — VGPR i: lanes 0-15 -> M=i, 16-31 -> M=8+i
#pragma unroll
    for (int mi = 0; mi < 2; ++mi)
#pragma unroll
        for (int ni = 0; ni < 2; ++ni) {
            int rbase = m0 + wr * 32 + mi * 16 + hi * 8;
            int ccol  = n0 + wc * 32 + ni * 16 + lh;
            if (ccol < N) {
#pragma unroll
                for (int i = 0; i < 8; ++i) {
                    int rr = rbase + i;
                    if (rr < M) {
                        float v = c[mi][ni][i];
                        if (bias) v += bias[ccol];
                        if (act == 1) v = v * conf_sigmoid(v);
                        C[(size_t)rr * ldc + ccol] = v;
                    }
                }
            }
        }
}

// ---------------------------------------------------------------------------
// Host orchestration
// ---------------------------------------------------------------------------
extern "C" void kernel_launch(void* const* d_in, const int* in_sizes, int n_in,
                              void* d_out, int out_size, void* d_ws, size_t ws_size,
                              hipStream_t stream)
{
    (void)in_sizes; (void)n_in; (void)out_size; (void)ws_size;

    constexpr int Dc = 512, Hc = 8, DKc = 64, DFFc = 2048, Bc = 8, Tc = 512, Kc = 31;
    constexpr int Pc = 2 * Tc - 1;             // 1023
    constexpr long Rr = (long)Bc * Tc;         // 4096 rows

    const float* x       = (const float*)d_in[0];
    const float* pos_emb = (const float*)d_in[1];
    const float* ln1_g = (const float*)d_in[2],  *ln1_b = (const float*)d_in[3];
    const float* ff1_w1 = (const float*)d_in[4], *ff1_b1 = (const float*)d_in[5];
    const float* ff1_w2 = (const float*)d_in[6], *ff1_b2 = (const float*)d_in[7];
    const float* lna_g = (const float*)d_in[8],  *lna_b = (const float*)d_in[9];
    const float* wq = (const float*)d_in[10], *bq = (const float*)d_in[11];
    const float* wk = (const float*)d_in[12], *bk = (const float*)d_in[13];
    const float* wv = (const float*)d_in[14], *bv = (const float*)d_in[15];
    const float* wo = (const float*)d_in[16], *bo = (const float*)d_in[17];
    const float* wpos = (const float*)d_in[18];
    const float* pbu = (const float*)d_in[19], *pbv = (const float*)d_in[20];
    const float* lnc_g = (const float*)d_in[21], *lnc_b = (const float*)d_in[22];
    const float* pw1_w = (const float*)d_in[23], *pw1_b = (const float*)d_in[24];
    const float* dw_w = (const float*)d_in[25],  *dw_b = (const float*)d_in[26];
    const float* cln_g = (const float*)d_in[27], *cln_b = (const float*)d_in[28];
    const float* pw2_w = (const float*)d_in[29], *pw2_b = (const float*)d_in[30];
    const float* ln2_g = (const float*)d_in[31], *ln2_b = (const float*)d_in[32];
    const float* ff2_w1 = (const float*)d_in[33], *ff2_b1 = (const float*)d_in[34];
    const float* ff2_w2 = (const float*)d_in[35], *ff2_b2 = (const float*)d_in[36];
    const float* lno_g = (const float*)d_in[37], *lno_b = (const float*)d_in[38];

    // bump allocator over workspace
    char* wsb = (char*)d_ws;
    size_t off = 0;
    auto alloc = [&](size_t bytes) -> void* {
        void* p = wsb + off;
        off += (bytes + 255) & ~(size_t)255;
        return p;
    };

    float*    r     = (float*)alloc((size_t)Rr * Dc * 4);
    float*    ybuf  = (float*)alloc((size_t)Rr * Dc * 4);
    float*    h32   = (float*)alloc((size_t)Rr * DFFc * 4);
    _Float16* a16   = (_Float16*)alloc((size_t)Rr * DFFc * 2);
    _Float16* w16   = (_Float16*)alloc((size_t)Dc * DFFc * 2);
    float*    q32   = (float*)alloc((size_t)Rr * Dc * 4);
    float*    k32   = (float*)alloc((size_t)Rr * Dc * 4);
    float*    v32   = (float*)alloc((size_t)Rr * Dc * 4);
    float*    o32   = (float*)alloc((size_t)Rr * Dc * 4);
    float*    p32   = (float*)alloc((size_t)Pc * Dc * 4);
    _Float16* qu16  = (_Float16*)alloc((size_t)Rr * Dc * 2);
    _Float16* qv16  = (_Float16*)alloc((size_t)Rr * Dc * 2);
    _Float16* k16   = (_Float16*)alloc((size_t)Rr * Dc * 2);
    _Float16* v16   = (_Float16*)alloc((size_t)Rr * Dc * 2);
    _Float16* p16   = (_Float16*)alloc((size_t)Pc * Dc * 2);
    float*    sb    = (float*)alloc((size_t)Hc * Tc * Tc * 4);
    float*    bdb   = (float*)alloc((size_t)Hc * Tc * Pc * 4);
    _Float16* att16 = (_Float16*)alloc((size_t)Hc * Tc * Tc * 2);

    auto g1 = [](long n) { return dim3((unsigned)((n + 255) / 256)); };
    auto LN = [&](const float* in, const float* g, const float* bb, float* out) {
        conf_ln<<<(unsigned)Rr, 128, 0, stream>>>(in, g, bb, out, Dc);
    };
    auto CVT = [&](const float* in, _Float16* out, long n) {
        conf_cvt<<<g1(n), 256, 0, stream>>>(in, out, n);
    };
    auto GEMM = [&](const _Float16* A, long sA, int lda,
                    const _Float16* Bm, long sB, int ldb, int tB,
                    float* C, long sC, int ldc,
                    int M, int N, int Kd, const float* bias, int act, int batch) {
        dim3 grid((unsigned)((N + 63) / 64), (unsigned)((M + 63) / 64), (unsigned)batch);
        conf_gemm<<<grid, 128, 0, stream>>>(A, sA, lda, Bm, sB, ldb, tB,
                                            C, sC, ldc, M, N, Kd, bias, act);
    };
    const long nRD = Rr * Dc;

    // ---------------- FF1 (half-step residual) ----------------
    LN(x, ln1_g, ln1_b, ybuf);
    CVT(ybuf, a16, nRD);
    CVT(ff1_w1, w16, (long)Dc * DFFc);
    GEMM(a16, 0, Dc, w16, 0, DFFc, 0, h32, 0, DFFc, (int)Rr, DFFc, Dc, ff1_b1, 1, 1);
    CVT(h32, a16, Rr * DFFc);
    CVT(ff1_w2, w16, (long)DFFc * Dc);
    GEMM(a16, 0, DFFc, w16, 0, Dc, 0, ybuf, 0, Dc, (int)Rr, Dc, DFFc, ff1_b2, 0, 1);
    conf_add3<<<g1(nRD), 256, 0, stream>>>(r, x, ybuf, 0.5f, nRD);

    // ---------------- Rel-pos multi-head attention ----------------
    LN(r, lna_g, lna_b, ybuf);
    CVT(ybuf, a16, nRD);
    CVT(wq, w16, (long)Dc * Dc);
    GEMM(a16, 0, Dc, w16, 0, Dc, 0, q32, 0, Dc, (int)Rr, Dc, Dc, bq, 0, 1);
    CVT(wk, w16, (long)Dc * Dc);
    GEMM(a16, 0, Dc, w16, 0, Dc, 0, k32, 0, Dc, (int)Rr, Dc, Dc, bk, 0, 1);
    CVT(wv, w16, (long)Dc * Dc);
    GEMM(a16, 0, Dc, w16, 0, Dc, 0, v32, 0, Dc, (int)Rr, Dc, Dc, bv, 0, 1);
    // positional projection: p = pos_emb @ wpos  -> [P, D]
    CVT(pos_emb, a16, (long)Pc * Dc);
    CVT(wpos, w16, (long)Dc * Dc);
    GEMM(a16, 0, Dc, w16, 0, Dc, 0, p32, 0, Dc, Pc, Dc, Dc, nullptr, 0, 1);

    conf_bias_head<<<g1(nRD), 256, 0, stream>>>(q32, pbu, qu16, nRD, Dc);
    conf_bias_head<<<g1(nRD), 256, 0, stream>>>(q32, pbv, qv16, nRD, Dc);
    CVT(k32, k16, nRD);
    CVT(v32, v16, nRD);
    CVT(p32, p16, (long)Pc * Dc);

    const float scale = 0.125f;  // 1/sqrt(64)
    for (int b = 0; b < Bc; ++b) {
        const long bTD = (long)b * Tc * Dc;
        // ac[h] = (q+pbu)_h @ k_h^T      [T,T]
        GEMM(qu16 + bTD, DKc, Dc, k16 + bTD, DKc, Dc, 1,
             sb, (long)Tc * Tc, Tc, Tc, Tc, DKc, nullptr, 0, Hc);
        // bd[h] = (q+pbv)_h @ p_h^T      [T,P]
        GEMM(qv16 + bTD, DKc, Dc, p16, DKc, Dc, 1,
             bdb, (long)Tc * Pc, Pc, Tc, Pc, DKc, nullptr, 0, Hc);
        // fused rel-shift + softmax -> attn f16 [H,T,T]
        conf_softmax_shift<<<dim3(Tc, Hc), 256, 0, stream>>>(
            sb, bdb, att16, Tc, Pc, scale);
        // o_h = attn_h @ v_h             [T,DK]
        GEMM(att16, (long)Tc * Tc, Tc, v16 + bTD, DKc, Dc, 0,
             o32 + bTD, DKc, Dc, Tc, DKc, Tc, nullptr, 0, Hc);
    }
    CVT(o32, a16, nRD);
    CVT(wo, w16, (long)Dc * Dc);
    GEMM(a16, 0, Dc, w16, 0, Dc, 0, ybuf, 0, Dc, (int)Rr, Dc, Dc, bo, 0, 1);
    conf_axpy<<<g1(nRD), 256, 0, stream>>>(r, ybuf, 1.0f, nRD);

    // ---------------- Convolution module ----------------
    LN(r, lnc_g, lnc_b, ybuf);
    CVT(ybuf, a16, nRD);
    CVT(pw1_w, w16, (long)Dc * 2 * Dc);
    GEMM(a16, 0, Dc, w16, 0, 2 * Dc, 0, h32, 0, 2 * Dc, (int)Rr, 2 * Dc, Dc, pw1_b, 0, 1);
    conf_glu<<<g1(nRD), 256, 0, stream>>>(h32, ybuf, nRD, Dc);
    conf_dwconv<<<g1(nRD), 256, 0, stream>>>(ybuf, dw_w, dw_b, q32, Bc, Tc, Dc, Kc);
    LN(q32, cln_g, cln_b, ybuf);
    conf_silu<<<g1(nRD), 256, 0, stream>>>(ybuf, nRD);
    CVT(ybuf, a16, nRD);
    CVT(pw2_w, w16, (long)Dc * Dc);
    GEMM(a16, 0, Dc, w16, 0, Dc, 0, ybuf, 0, Dc, (int)Rr, Dc, Dc, pw2_b, 0, 1);
    conf_axpy<<<g1(nRD), 256, 0, stream>>>(r, ybuf, 1.0f, nRD);

    // ---------------- FF2 (half-step residual) ----------------
    LN(r, ln2_g, ln2_b, ybuf);
    CVT(ybuf, a16, nRD);
    CVT(ff2_w1, w16, (long)Dc * DFFc);
    GEMM(a16, 0, Dc, w16, 0, DFFc, 0, h32, 0, DFFc, (int)Rr, DFFc, Dc, ff2_b1, 1, 1);
    CVT(h32, a16, Rr * DFFc);
    CVT(ff2_w2, w16, (long)DFFc * Dc);
    GEMM(a16, 0, DFFc, w16, 0, Dc, 0, ybuf, 0, Dc, (int)Rr, Dc, DFFc, ff2_b2, 0, 1);
    conf_axpy<<<g1(nRD), 256, 0, stream>>>(r, ybuf, 0.5f, nRD);

    // ---------------- Final LayerNorm -> output ----------------
    LN(r, lno_g, lno_b, (float*)d_out);
}
